// SimpleGAT_36747740184681
// MI455X (gfx1250) — matrix operand order repaired and checked by reading, not compiled, not run on hardware
//
#include <hip/hip_runtime.h>
#include <stdint.h>

#define NEG_SLOPE 0.2f
#define EPSV 1e-16f
#define CDIV(a, b) (((a) + (b) - 1) / (b))

typedef __attribute__((ext_vector_type(16))) __bf16 v16bf;
typedef __attribute__((ext_vector_type(8)))  float  v8f;

union Frag16 { v16bf v; unsigned int u[8]; };

// ---------------- utility kernels ----------------

__global__ void k_fill_f32(float* __restrict__ p, float v, int n) {
    int i = blockIdx.x * blockDim.x + threadIdx.x;
    if (i < n) p[i] = v;
}

__device__ __forceinline__ unsigned short f32_to_bf16_rne(float f) {
    unsigned int u = __float_as_uint(f);
    unsigned int r = u + 0x7FFFu + ((u >> 16) & 1u);   // round to nearest even
    return (unsigned short)(r >> 16);
}

__global__ void k_f32_to_bf16(const float* __restrict__ s, unsigned short* __restrict__ d, int n) {
    int i = blockIdx.x * blockDim.x + threadIdx.x;
    if (i < n) d[i] = f32_to_bf16_rne(s[i]);
}

// ---------------- WMMA GEMM:  C[M x NC] = A[M x K] * B[NC x K]^T ----------------
// grid.x = M/16 row tiles; threadIdx.y = column tile within block (NC/16 waves).
// Per-lane fragment packing follows ISA 7.12.2 (wave32, 16-bit A 16x32 / B 32x16).
template <int K>
__global__ void k_gemm_wmma_bf16(const unsigned short* __restrict__ A,
                                 const unsigned short* __restrict__ B,
                                 float* __restrict__ C, int NC) {
    const int lane = threadIdx.x;          // 0..31
    const int half = lane >> 4;            // 0 or 1
    const int l15  = lane & 15;
    const int row0 = blockIdx.x * 16;
    const int col0 = threadIdx.y * 16;

    const unsigned short* arow = A + (size_t)(row0 + l15) * K;  // A row for M=l15
    const unsigned short* brow = B + (size_t)(col0 + l15) * K;  // W row for N=l15

    v8f acc = {0.f, 0.f, 0.f, 0.f, 0.f, 0.f, 0.f, 0.f};
#pragma unroll
    for (int k0 = 0; k0 < K; k0 += 32) {
        Frag16 fa, fb;
#pragma unroll
        for (int j = 0; j < 4; ++j) {   // A: VGPR j -> K = 8*half + 2j (+1); VGPR 4+j -> +16
            fa.u[j]     = *(const unsigned int*)(arow + k0 + 8 * half + 2 * j);
            fa.u[4 + j] = *(const unsigned int*)(arow + k0 + 16 + 8 * half + 2 * j);
        }
#pragma unroll
        for (int j = 0; j < 4; ++j) {   // B: VGPR j -> K = 8*half + 2j (+1); VGPR 4+j -> +16
            fb.u[j]     = *(const unsigned int*)(brow + k0 + 8 * half + 2 * j);
            fb.u[4 + j] = *(const unsigned int*)(brow + k0 + 16 + 8 * half + 2 * j);
        }
        acc = __builtin_amdgcn_wmma_f32_16x16x32_bf16(false, fa.v, false, fb.v,
                                                      (short)0, acc, false, false);
    }
#pragma unroll
    for (int j = 0; j < 8; ++j) {       // D: VGPR j -> row 8*half + j, col l15
        C[(size_t)(row0 + 8 * half + j) * NC + col0 + l15] = acc[j];
    }
}

// ---------------- attention logits: a_s[n,h] = <xh[n,h,:], att_s[h,:]> ----------------
__global__ void k_att_scores(const float* __restrict__ xh,
                             const float* __restrict__ att_s,
                             const float* __restrict__ att_d,
                             float* __restrict__ as_, float* __restrict__ ad_,
                             int N, int H, int C) {
    int i = blockIdx.x * blockDim.x + threadIdx.x;
    if (i >= N * H) return;
    int h = i % H, n = i / H;
    const float* xr = xh + (size_t)n * H * C + (size_t)h * C;
    const float* s  = att_s + h * C;
    const float* d  = att_d + h * C;
    float ss = 0.f, sd = 0.f;
    for (int c = 0; c < C; ++c) { float v = xr[c]; ss += v * s[c]; sd += v * d[c]; }
    as_[i] = ss; ad_[i] = sd;
}

// float atomic-max via signed-max / unsigned-min trick (monotone over IEEE bits)
__device__ __forceinline__ void atomicMaxF(float* addr, float v) {
    if (v >= 0.f) atomicMax((int*)addr, __float_as_int(v));
    else          atomicMin((unsigned int*)addr, __float_as_uint(v));
}

// ---------------- edge kernels (edges 0..E-1 real, E..E+N-1 self-loops) ----------------
__global__ void k_edge_max(const long long* __restrict__ src, const long long* __restrict__ dst,
                           const float* __restrict__ as_, const float* __restrict__ ad_,
                           float* __restrict__ m, int E, int Et, int H) {
    int i = blockIdx.x * blockDim.x + threadIdx.x;
    if (i >= Et * H) return;
    int e = i / H, h = i - e * H;
    int s = (e < E) ? (int)src[e] : (e - E);
    int d = (e < E) ? (int)dst[e] : (e - E);
    float a = as_[s * H + h] + ad_[d * H + h];
    a = (a < 0.f) ? NEG_SLOPE * a : a;
    atomicMaxF(&m[d * H + h], a);
}

__global__ void k_edge_exp(const long long* __restrict__ src, const long long* __restrict__ dst,
                           const float* __restrict__ as_, const float* __restrict__ ad_,
                           const float* __restrict__ m, float* __restrict__ expv,
                           float* __restrict__ denom, int E, int Et, int H) {
    int i = blockIdx.x * blockDim.x + threadIdx.x;
    if (i >= Et * H) return;
    int e = i / H, h = i - e * H;
    int s = (e < E) ? (int)src[e] : (e - E);
    int d = (e < E) ? (int)dst[e] : (e - E);
    float a = as_[s * H + h] + ad_[d * H + h];
    a = (a < 0.f) ? NEG_SLOPE * a : a;
    float ev = __expf(a - m[d * H + h]);
    expv[i] = ev;
    atomicAdd(&denom[d * H + h], ev);
}

// one block per edge; threads cover H*C contiguous channels -> coalesced gather + atomics
__global__ void k_edge_msg(const long long* __restrict__ src, const long long* __restrict__ dst,
                           const float* __restrict__ xh, const float* __restrict__ expv,
                           const float* __restrict__ denom, float* __restrict__ acc,
                           int E, int H, int C) {
    int e = blockIdx.x;
    int c = threadIdx.x;                 // 0 .. H*C-1
    int h = c / C;
    int s = (e < E) ? (int)src[e] : (e - E);
    int d = (e < E) ? (int)dst[e] : (e - E);
    float w = expv[e * H + h] / (denom[d * H + h] + EPSV);
    atomicAdd(&acc[(size_t)d * H * C + c], xh[(size_t)s * H * C + c] * w);
}

// layer-1 finalize: bias + relu, emit bf16 for the next WMMA GEMM
__global__ void k_bias_relu_bf16(const float* __restrict__ acc, const float* __restrict__ bias,
                                 unsigned short* __restrict__ o, int N, int C) {
    int i = blockIdx.x * blockDim.x + threadIdx.x;
    if (i >= N * C) return;
    float v = acc[i] + bias[i % C];
    o[i] = f32_to_bf16_rne(fmaxf(v, 0.f));
}

// layer-2 finalize: bias + L2 normalize (one wave32 per node, C=64 -> 2 vals/lane)
__global__ void k_finalize_norm(const float* __restrict__ acc, const float* __restrict__ bias,
                                float* __restrict__ out, int C) {
    int n = blockIdx.x;
    int l = threadIdx.x;  // 0..31
    float v0 = acc[(size_t)n * C + l]      + bias[l];
    float v1 = acc[(size_t)n * C + l + 32] + bias[l + 32];
    float ss = v0 * v0 + v1 * v1;
#pragma unroll
    for (int o = 16; o > 0; o >>= 1) ss += __shfl_xor(ss, o, 32);
    float inv = 1.f / fmaxf(sqrtf(ss), 1e-12f);
    out[(size_t)n * C + l]      = v0 * inv;
    out[(size_t)n * C + l + 32] = v1 * inv;
}

// ---------------- launch ----------------
extern "C" void kernel_launch(void* const* d_in, const int* in_sizes, int n_in,
                              void* d_out, int out_size, void* d_ws, size_t ws_size,
                              hipStream_t stream) {
    const float*     x      = (const float*)d_in[0];
    const long long* ei     = (const long long*)d_in[1];   // int64 edge_index [2,E]
    const float*     W1     = (const float*)d_in[2];       // [128,128]
    const float*     att_s1 = (const float*)d_in[3];       // [1,2,64]
    const float*     att_d1 = (const float*)d_in[4];
    const float*     b1     = (const float*)d_in[5];       // [128]
    const float*     W2     = (const float*)d_in[6];       // [64,128]
    const float*     att_s2 = (const float*)d_in[7];       // [1,1,64]
    const float*     att_d2 = (const float*)d_in[8];
    const float*     b2     = (const float*)d_in[9];       // [64]
    float* out = (float*)d_out;

    const int N  = in_sizes[0] / 128;    // 50000 (multiple of 16)
    const int E  = in_sizes[1] / 2;      // 800000
    const int Et = E + N;                // + self loops
    const long long* srcE = ei;
    const long long* dstE = ei + E;

    // workspace carve (256B aligned)
    char* p = (char*)d_ws;
    auto alloc = [&](size_t bytes) -> void* {
        void* r = (void*)p;
        p += (bytes + 255) & ~(size_t)255;
        return r;
    };
    unsigned short* xb   = (unsigned short*)alloc((size_t)N * 128 * 2);
    unsigned short* w1b  = (unsigned short*)alloc(128 * 128 * 2);
    unsigned short* w2b  = (unsigned short*)alloc(64 * 128 * 2);
    unsigned short* hb   = (unsigned short*)alloc((size_t)N * 128 * 2);
    float* xh1  = (float*)alloc((size_t)N * 128 * 4);
    float* xh2  = (float*)alloc((size_t)N * 64 * 4);
    float* as1  = (float*)alloc((size_t)N * 2 * 4);
    float* ad1  = (float*)alloc((size_t)N * 2 * 4);
    float* m1   = (float*)alloc((size_t)N * 2 * 4);
    float* den1 = (float*)alloc((size_t)N * 2 * 4);
    float* as2  = (float*)alloc((size_t)N * 4);
    float* ad2  = (float*)alloc((size_t)N * 4);
    float* m2   = (float*)alloc((size_t)N * 4);
    float* den2 = (float*)alloc((size_t)N * 4);
    float* expv = (float*)alloc((size_t)Et * 2 * 4);       // reused by layer 2 (H=1)
    float* acc1 = (float*)alloc((size_t)N * 128 * 4);
    float* acc2 = (float*)alloc((size_t)N * 64 * 4);

    const int NB = 256;

    // bf16 staging
    k_f32_to_bf16<<<CDIV(N * 128, NB), NB, 0, stream>>>(x, xb, N * 128);
    k_f32_to_bf16<<<CDIV(128 * 128, NB), NB, 0, stream>>>(W1, w1b, 128 * 128);
    k_f32_to_bf16<<<CDIV(64 * 128, NB), NB, 0, stream>>>(W2, w2b, 64 * 128);

    // ---- layer 1 (H=2, C=64) ----
    k_gemm_wmma_bf16<128><<<dim3(N / 16), dim3(32, 8), 0, stream>>>(xb, w1b, xh1, 128);
    k_att_scores<<<CDIV(N * 2, NB), NB, 0, stream>>>(xh1, att_s1, att_d1, as1, ad1, N, 2, 64);
    k_fill_f32<<<CDIV(N * 2, NB), NB, 0, stream>>>(m1, -1e30f, N * 2);
    k_fill_f32<<<CDIV(N * 2, NB), NB, 0, stream>>>(den1, 0.f, N * 2);
    k_fill_f32<<<CDIV(N * 128, NB), NB, 0, stream>>>(acc1, 0.f, N * 128);
    k_edge_max<<<CDIV(Et * 2, NB), NB, 0, stream>>>(srcE, dstE, as1, ad1, m1, E, Et, 2);
    k_edge_exp<<<CDIV(Et * 2, NB), NB, 0, stream>>>(srcE, dstE, as1, ad1, m1, expv, den1, E, Et, 2);
    k_edge_msg<<<dim3(Et), dim3(128), 0, stream>>>(srcE, dstE, xh1, expv, den1, acc1, E, 2, 64);
    k_bias_relu_bf16<<<CDIV(N * 128, NB), NB, 0, stream>>>(acc1, b1, hb, N, 128);

    // ---- layer 2 (H=1, C=64) ----
    k_gemm_wmma_bf16<128><<<dim3(N / 16), dim3(32, 4), 0, stream>>>(hb, w2b, xh2, 64);
    k_att_scores<<<CDIV(N, NB), NB, 0, stream>>>(xh2, att_s2, att_d2, as2, ad2, N, 1, 64);
    k_fill_f32<<<CDIV(N, NB), NB, 0, stream>>>(m2, -1e30f, N);
    k_fill_f32<<<CDIV(N, NB), NB, 0, stream>>>(den2, 0.f, N);
    k_fill_f32<<<CDIV(N * 64, NB), NB, 0, stream>>>(acc2, 0.f, N * 64);
    k_edge_max<<<CDIV(Et, NB), NB, 0, stream>>>(srcE, dstE, as2, ad2, m2, E, Et, 1);
    k_edge_exp<<<CDIV(Et, NB), NB, 0, stream>>>(srcE, dstE, as2, ad2, m2, expv, den2, E, Et, 1);
    k_edge_msg<<<dim3(Et), dim3(64), 0, stream>>>(srcE, dstE, xh2, expv, den2, acc2, E, 1, 64);
    k_finalize_norm<<<dim3(N), dim3(32), 0, stream>>>(acc2, b2, out, 64);

    (void)n_in; (void)out_size; (void)ws_size;
}
